// DecoderBlock_66718021976788
// MI455X (gfx1250) — matrix operand order repaired and checked by reading, not compiled, not run on hardware
//
#include <hip/hip_runtime.h>

// ---------------------------------------------------------------------------
// CDNA5 / gfx1250 decoder block.
// Linear (softmax-free) attention refactored via associativity:
//   out = q (k^T v) * scale^-0.5 ; out @ Wm = q @ (kv @ Wm_h stacked)
// All GEMMs run on V_WMMA_F32_16X16X32_BF16 (wave32), fp32 accumulation.
// Tiles staged through LDS with b128 global loads; B tile stored transposed
// so both A and B fragments load as contiguous ds_load_b128 pairs.
// ---------------------------------------------------------------------------

typedef __attribute__((ext_vector_type(16))) __bf16 v16bf;
typedef __attribute__((ext_vector_type(8)))  float  v8f;

__device__ __forceinline__ unsigned short f2bf(float f) {
  unsigned int u = __float_as_uint(f);
  u += 0x7FFFu + ((u >> 16) & 1u);          // round-to-nearest-even
  return (unsigned short)(u >> 16);
}

// ---------------------------------------------------------------------------
// fp32 -> bf16 conversion (weights)
// ---------------------------------------------------------------------------
__global__ void cvt_f2bf_kernel(const float* __restrict__ s,
                                unsigned short* __restrict__ d, long long n) {
  long long i = (long long)blockIdx.x * blockDim.x + threadIdx.x;
  long long stride = (long long)gridDim.x * blockDim.x;
  for (; i < n; i += stride) d[i] = f2bf(s[i]);
}

// ---------------------------------------------------------------------------
// Row LayerNorm (+optional pos add, +optional swish) -> bf16
// one block (256 threads) per row
// ---------------------------------------------------------------------------
__global__ void __launch_bounds__(256)
ln_act_kernel(const float* __restrict__ X, int width,
              const float* __restrict__ g, const float* __restrict__ b,
              const float* __restrict__ pos, int posMod,
              int doSwish, unsigned short* __restrict__ outB) {
  __shared__ float s1[256];
  __shared__ float s2[256];
  const long long row = blockIdx.x;
  const float* xr = X + row * width;
  float sum = 0.f, sq = 0.f;
  for (int i = threadIdx.x; i < width; i += 256) {
    float v = xr[i]; sum += v; sq += v * v;
  }
  s1[threadIdx.x] = sum; s2[threadIdx.x] = sq;
  __syncthreads();
  for (int s = 128; s > 0; s >>= 1) {
    if ((int)threadIdx.x < s) {
      s1[threadIdx.x] += s1[threadIdx.x + s];
      s2[threadIdx.x] += s2[threadIdx.x + s];
    }
    __syncthreads();
  }
  const float mu  = s1[0] / (float)width;
  const float var = s2[0] / (float)width - mu * mu;
  const float inv = rsqrtf(var + 1e-5f);
  const float* pr = pos ? pos + (long long)((int)row % posMod) * width : nullptr;
  unsigned short* orow = outB + row * width;
  for (int i = threadIdx.x; i < width; i += 256) {
    float v = (xr[i] - mu) * inv * g[i] + b[i];
    if (pr) v += pr[i];
    if (doSwish) v = v / (1.f + __expf(-v));
    orow[i] = f2bf(v);
  }
}

// ---------------------------------------------------------------------------
// Generic batched bf16 GEMM, fp32 accumulation via v_wmma_f32_16x16x32_bf16.
//   D = alpha * (A @ B) + bias + residual
// Block = 128 threads (4 wave32), 64x64 tile, each wave 32x32 (2x2 wmma).
// Batch offset = (z/innerDiv)*outer_stride + (z%innerDiv)*inner_stride.
// transA: A stored K-major, A[m,k] = Aptr[k*lda + m].
// Alignment contract (all call sites satisfy): lda/ldb multiples of 8,
// tile origins multiples of 64, base pointers 16B-aligned.
// ---------------------------------------------------------------------------
__global__ void __launch_bounds__(128)
gemm_bf16_kernel(const unsigned short* __restrict__ A,
                 long long Aos, long long Ais, int lda, int transA,
                 const unsigned short* __restrict__ Bm,
                 long long Bos, long long Bis, int ldb,
                 const float* __restrict__ bias,
                 const float* __restrict__ res,
                 long long Ros, long long Ris, int ldr,
                 const float* __restrict__ alphaPtr, float alphaC,
                 float* __restrict__ outF,
                 unsigned short* __restrict__ outB,
                 long long Oos, long long Ois, int ldo,
                 int M, int N, int K, int innerDiv) {
  __shared__ unsigned short As[64][40];   // [m][k]   row stride 80B (16B aligned)
  __shared__ unsigned short Bs[64][40];   // [n][k]   transposed-in-LDS B tile

  const int tid   = threadIdx.x;
  const int lane  = tid & 31;
  const int wave  = tid >> 5;
  const int warpM = wave >> 1;            // 0..1
  const int warpN = wave & 1;             // 0..1
  const int half  = lane >> 4;            // 0..1
  const int l16   = lane & 15;

  const long long z  = blockIdx.z;
  const long long zo = z / innerDiv;
  const long long zi = z % innerDiv;
  const unsigned short* Ab = A  + zo * Aos + zi * Ais;
  const unsigned short* Bb = Bm + zo * Bos + zi * Bis;

  const int tm = blockIdx.y * 64;
  const int tn = blockIdx.x * 64;

  v8f acc[2][2] = {};

  const int ksteps = K >> 5;
  for (int kt = 0; kt < ksteps; ++kt) {
    const int kb = kt << 5;

    // ---- stage A tile (64 m x 32 k) via b128 chunks ----
    if (!transA) {
#pragma unroll
      for (int c = 0; c < 2; ++c) {
        int q = c * 128 + tid;            // 256 chunks of 8 elems
        int m = q >> 2, ko = (q & 3) << 3;
        int4 v = *(const int4*)(Ab + (long long)(tm + m) * lda + (kb + ko));
        *(int4*)&As[m][ko] = v;           // LDS b128 store (16B aligned)
      }
    } else {
#pragma unroll
      for (int c = 0; c < 2; ++c) {
        int q = c * 128 + tid;
        int kk = q >> 3, mo = (q & 7) << 3;   // 8 contiguous m's
        int4 v = *(const int4*)(Ab + (long long)(kb + kk) * lda + (tm + mo));
        const unsigned short* pv = (const unsigned short*)&v;
#pragma unroll
        for (int j = 0; j < 8; ++j) As[mo + j][kk] = pv[j];
      }
    }
    // ---- stage B tile (32 k x 64 n) transposed into Bs[n][k] ----
#pragma unroll
    for (int c = 0; c < 2; ++c) {
      int q = c * 128 + tid;
      int kk = q >> 3, no = (q & 7) << 3;     // 8 contiguous n's
      int4 v = *(const int4*)(Bb + (long long)(kb + kk) * ldb + (tn + no));
      const unsigned short* pv = (const unsigned short*)&v;
#pragma unroll
      for (int j = 0; j < 8; ++j) Bs[no + j][kk] = pv[j];
    }
    if (kt + 1 < ksteps) {  // global_prefetch_b8 of next K tile
      int nkb = kb + 32;
      __builtin_prefetch(Bb + (long long)(nkb + (tid >> 2)) * ldb + tn, 0, 1);
      if (!transA)
        __builtin_prefetch(Ab + (long long)(tm + (tid >> 1)) * lda + nkb, 0, 1);
      else
        __builtin_prefetch(Ab + (long long)nkb * lda + tm + (tid & 63), 0, 1);
    }
    __syncthreads();

    // ---- fragments (contiguous u32 reads -> ds_load_b128 pairs) ----
    union UF { v16bf v; unsigned int w[8]; } aF[2], bF[2];
#pragma unroll
    for (int fm = 0; fm < 2; ++fm) {
      int row = warpM * 32 + fm * 16 + l16;               // M = lane%16
#pragma unroll
      for (int j = 0; j < 8; ++j) {
        int kkb = (j >> 2) * 16 + half * 8 + (j & 3) * 2; // A 16-bit layout
        aF[fm].w[j] = *(const unsigned int*)&As[row][kkb];
      }
    }
#pragma unroll
    for (int fn = 0; fn < 2; ++fn) {
      int col = warpN * 32 + fn * 16 + l16;               // N = lane%16
#pragma unroll
      for (int j = 0; j < 8; ++j) {
        int kk = (j >> 2) * 16 + half * 8 + (j & 3) * 2;  // B 16-bit layout
        bF[fn].w[j] = *(const unsigned int*)&Bs[col][kk];
      }
    }
#pragma unroll
    for (int fm = 0; fm < 2; ++fm)
#pragma unroll
      for (int fn = 0; fn < 2; ++fn)
        acc[fm][fn] = __builtin_amdgcn_wmma_f32_16x16x32_bf16(
            false, aF[fm].v, false, bF[fn].v, (short)0, acc[fm][fn],
            false, false);
    __syncthreads();
  }

  float alpha = alphaC;
  if (alphaPtr) alpha *= rsqrtf(*alphaPtr);

  const float* Rb = res ? res + zo * Ros + zi * Ris : nullptr;
  const long long oOff = zo * Oos + zi * Ois;

#pragma unroll
  for (int fm = 0; fm < 2; ++fm) {
#pragma unroll
    for (int fn = 0; fn < 2; ++fn) {
      const int gn = tn + warpN * 32 + fn * 16 + l16;
#pragma unroll
      for (int r = 0; r < 8; ++r) {
        const int gm = tm + warpM * 32 + fm * 16 + half * 8 + r;
        float v = acc[fm][fn][r] * alpha;
        if (bias) v += bias[gn];
        if (Rb)   v += Rb[(long long)gm * ldr + gn];
        const long long o = oOff + (long long)gm * ldo + gn;
        if (outF) outF[o] = v;
        if (outB) outB[o] = f2bf(v);
      }
    }
  }
}

// ---------------------------------------------------------------------------
// host launcher
// ---------------------------------------------------------------------------
extern "C" void kernel_launch(void* const* d_in, const int* in_sizes, int n_in,
                              void* d_out, int out_size, void* d_ws, size_t ws_size,
                              hipStream_t stream) {
  (void)in_sizes; (void)n_in; (void)out_size; (void)ws_size;
  const int B = 8, N = 1024, D = 512, H = 8, DE = 2048;
  const int QKV = 5120;                 // 512 + 512 + 4096
  const long long Rows = (long long)B * N;  // 8192

  const float* x       = (const float*)d_in[0];
  const float* pos     = (const float*)d_in[1];
  const float* scaleP  = (const float*)d_in[2];
  const float* ln0_g   = (const float*)d_in[3];
  const float* ln0_b   = (const float*)d_in[4];
  const float* W_qkv   = (const float*)d_in[5];
  const float* b_qkv   = (const float*)d_in[6];
  const float* W_merge = (const float*)d_in[7];
  const float* b_merge = (const float*)d_in[8];
  const float* ln1_g   = (const float*)d_in[9];
  const float* ln1_b   = (const float*)d_in[10];
  const float* W_ff1   = (const float*)d_in[11];
  const float* b_ff1   = (const float*)d_in[12];
  const float* ln2_g   = (const float*)d_in[13];
  const float* ln2_b   = (const float*)d_in[14];
  const float* W_ff2   = (const float*)d_in[15];
  const float* b_ff2   = (const float*)d_in[16];

  size_t off = 0;
  auto take = [&](size_t bytes) -> void* {
    void* p = (char*)d_ws + off;
    off = (off + bytes + 255) & ~(size_t)255;
    return p;
  };
  unsigned short* Wqkv_bf = (unsigned short*)take((size_t)D * QKV * 2);
  unsigned short* Wm_bf   = (unsigned short*)take((size_t)4096 * D * 2);
  unsigned short* Wf1_bf  = (unsigned short*)take((size_t)D * DE * 2);
  unsigned short* Wf2_bf  = (unsigned short*)take((size_t)DE * D * 2);
  unsigned short* hbf     = (unsigned short*)take((size_t)Rows * D * 2);
  unsigned short* qkv_bf  = (unsigned short*)take((size_t)Rows * QKV * 2);
  unsigned short* kv_bf   = (unsigned short*)take((size_t)B * H * 64 * D * 2);
  unsigned short* Mst_bf  = (unsigned short*)take((size_t)B * 512 * 512 * 2);
  float*          x1      = (float*)take((size_t)Rows * D * 4);
  unsigned short* ln1h    = (unsigned short*)take((size_t)Rows * D * 2);
  unsigned short* gbf     = (unsigned short*)take((size_t)Rows * DE * 2);
  float*          f1      = (float*)qkv_bf;   // qkv dead after attn GEMM; 64MB <= 80MB

  // 0) weight conversions fp32 -> bf16
  cvt_f2bf_kernel<<<2048, 256, 0, stream>>>(W_qkv,   Wqkv_bf, (long long)D * QKV);
  cvt_f2bf_kernel<<<2048, 256, 0, stream>>>(W_merge, Wm_bf,   (long long)4096 * D);
  cvt_f2bf_kernel<<<1024, 256, 0, stream>>>(W_ff1,   Wf1_bf,  (long long)D * DE);
  cvt_f2bf_kernel<<<1024, 256, 0, stream>>>(W_ff2,   Wf2_bf,  (long long)DE * D);

  // 1) h = swish(LN0(x) + pos) -> bf16
  ln_act_kernel<<<(int)Rows, 256, 0, stream>>>(x, D, ln0_g, ln0_b, pos, N, 1, hbf);

  // 2) qkv = h @ W_qkv (no bias)  -> bf16  (8192 x 5120, K=512)
  {
    dim3 g(QKV / 64, Rows / 64, 1);
    gemm_bf16_kernel<<<g, 128, 0, stream>>>(
        hbf, 0, 0, D, 0,
        Wqkv_bf, 0, 0, QKV,
        nullptr, nullptr, 0, 0, 0,
        nullptr, 1.f,
        nullptr, qkv_bf, 0, 0, QKV,
        (int)Rows, QKV, D, 1);
  }
  // 3) kv[b,h] = (k_h^T @ v_h) * scale^-0.5  -> bf16  (64 x 512, K=1024) x 64
  {
    dim3 g(D / 64, 1, B * H);
    gemm_bf16_kernel<<<g, 128, 0, stream>>>(
        qkv_bf + 512,  (long long)N * QKV, 64,  QKV, 1,   // A = k^T (per b, per h)
        qkv_bf + 1024, (long long)N * QKV, 512, QKV,      // B = v   (per b, per h)
        nullptr, nullptr, 0, 0, 0,
        scaleP, 1.f,
        nullptr, kv_bf, (long long)H * 64 * D, (long long)64 * D, D,
        64, D, N, H);
  }
  // 4) Mstack[b, h*64+m, :] = kv[b,h] @ W_merge_h  -> bf16  (64 x 512, K=512) x 64
  {
    dim3 g(D / 64, 1, B * H);
    gemm_bf16_kernel<<<g, 128, 0, stream>>>(
        kv_bf, (long long)H * 64 * D, (long long)64 * D, D, 0,
        Wm_bf, 0, (long long)512 * D, D,
        nullptr, nullptr, 0, 0, 0,
        nullptr, 1.f,
        nullptr, Mst_bf, (long long)512 * 512, (long long)64 * D, D,
        64, D, D, H);
  }
  // 5) x1 = x + q(b) @ Mstack(b) + b_merge  -> fp32  (1024 x 512, K=512) x 8
  {
    dim3 g(D / 64, N / 64, B);
    gemm_bf16_kernel<<<g, 128, 0, stream>>>(
        qkv_bf, 0, (long long)N * QKV, QKV, 0,
        Mst_bf, 0, (long long)512 * 512, D,
        b_merge,
        x, 0, (long long)N * D, D,
        nullptr, 1.f,
        x1, nullptr, 0, (long long)N * D, D,
        N, D, D, B);
  }
  // 6) ln1h = LN1(x1) -> bf16
  ln_act_kernel<<<(int)Rows, 256, 0, stream>>>(x1, D, ln1_g, ln1_b, nullptr, 1, 0, ln1h);

  // 7) f1 = ln1h @ W_ff1 + b_ff1 -> fp32  (8192 x 2048, K=512)
  {
    dim3 g(DE / 64, Rows / 64, 1);
    gemm_bf16_kernel<<<g, 128, 0, stream>>>(
        ln1h, 0, 0, D, 0,
        Wf1_bf, 0, 0, DE,
        b_ff1, nullptr, 0, 0, 0,
        nullptr, 1.f,
        f1, nullptr, 0, 0, DE,
        (int)Rows, DE, D, 1);
  }
  // 8) g = swish(LN2(f1)) -> bf16
  ln_act_kernel<<<(int)Rows, 256, 0, stream>>>(f1, DE, ln2_g, ln2_b, nullptr, 1, 1, gbf);

  // 9) out = g @ W_ff2 + b_ff2 + x1 -> fp32  (8192 x 512, K=2048)
  {
    dim3 g(D / 64, Rows / 64, 1);
    gemm_bf16_kernel<<<g, 128, 0, stream>>>(
        gbf, 0, 0, DE, 0,
        Wf2_bf, 0, 0, D,
        b_ff2,
        x1, 0, 0, D,
        nullptr, 1.f,
        (float*)d_out, nullptr, 0, 0, D,
        (int)Rows, D, DE, 1);
  }
}
